// GATWithSentenceEmbedding_17746804867563
// MI455X (gfx1250) — compile-verified, matrix operand order checked by reading, not run on hardware
//
#include <hip/hip_runtime.h>
#include <math.h>

typedef __attribute__((ext_vector_type(16))) _Float16 v16h;
typedef __attribute__((ext_vector_type(8)))  _Float16 v8h;
typedef __attribute__((ext_vector_type(8)))  float    v8f;

#define NEG_SLOPE 0.2f
#define GEPS 1e-16f

__device__ __forceinline__ unsigned f2ord(float f){
  unsigned u = __float_as_uint(f);
  return (u & 0x80000000u) ? ~u : (u | 0x80000000u);
}
__device__ __forceinline__ float ord2f(unsigned u){
  return (u & 0x80000000u) ? __uint_as_float(u & 0x7FFFFFFFu) : __uint_as_float(~u);
}

// sent[j] = fc0_w[j,:] . cls + fc0_b[j]     (1 block of 128 threads)
__global__ void k_sent(const float* __restrict__ fc0_w, const float* __restrict__ cls,
                       const float* __restrict__ fc0_b, float* __restrict__ sent){
  int j = threadIdx.x;
  const float* row = fc0_w + j * 768;
  float acc = 0.f;
  for (int k = 0; k < 768; ++k) acc += row[k] * cls[k];
  sent[j] = acc + fc0_b[j];
}

// c1[j] = sum_k sent[k] * W1[128+k, j]   (W1 is [256,128] row-major)
__global__ void k_c1(const float* __restrict__ W1, const float* __restrict__ sent,
                     float* __restrict__ c1){
  int j = threadIdx.x;
  float acc = 0.f;
  for (int k = 0; k < 128; ++k) acc += sent[k] * W1[(128 + k) * 128 + j];
  c1[j] = acc;
}

// dst[n*K + k] = (f16) src[k*Ncols + n]   (transpose + convert)
__global__ void k_transpose_f16(_Float16* __restrict__ dst, const float* __restrict__ src,
                                int K, int Ncols, int total){
  int idx = blockIdx.x * blockDim.x + threadIdx.x;
  if (idx >= total) return;
  int n = idx / K, k = idx % K;
  dst[n * K + k] = (_Float16)src[k * Ncols + n];
}

__global__ void k_f32_to_f16(_Float16* __restrict__ dst, const float* __restrict__ src, int total){
  int idx = blockIdx.x * blockDim.x + threadIdx.x;
  if (idx >= total) return;
  dst[idx] = (_Float16)src[idx];
}

// C[M,Nout] = A[M,K](f16) @ Bt[Nout,K](f16)^T  (+ optional per-column addvec)
// block = 256 thr (8 waves): 16 rows x 128 cols. grid(M/16, Nout/128).
__global__ void k_gemm(const _Float16* __restrict__ A, const _Float16* __restrict__ Bt,
                       float* __restrict__ C, int K, int Nout,
                       const float* __restrict__ addvec){
  int lane = threadIdx.x & 31;
  int wave = threadIdx.x >> 5;
  int m0 = blockIdx.x * 16;
  int n0 = blockIdx.y * 128 + wave * 16;
  int mr = lane & 15;
  int koff = (lane < 16) ? 0 : 8;
  const _Float16* arow = A + (size_t)(m0 + mr) * K;
  const _Float16* brow = Bt + (size_t)(n0 + mr) * K;
  v8f acc = {};
  for (int k0 = 0; k0 < K; k0 += 32){
    v8h alo = *(const v8h*)(arow + k0 + koff);
    v8h ahi = *(const v8h*)(arow + k0 + koff + 16);
    v8h blo = *(const v8h*)(brow + k0 + koff);
    v8h bhi = *(const v8h*)(brow + k0 + koff + 16);
    v16h a = __builtin_shufflevector(alo, ahi, 0,1,2,3,4,5,6,7,8,9,10,11,12,13,14,15);
    v16h b = __builtin_shufflevector(blo, bhi, 0,1,2,3,4,5,6,7,8,9,10,11,12,13,14,15);
    acc = __builtin_amdgcn_wmma_f32_16x16x32_f16(false, a, false, b, (short)0, acc, false, false);
  }
  int rowoff = (lane < 16) ? 0 : 8;
  float add = addvec ? addvec[n0 + mr] : 0.f;
  #pragma unroll
  for (int j = 0; j < 8; ++j)
    C[(size_t)(m0 + rowoff + j) * Nout + (n0 + mr)] = acc[j] + add;
}

// per (node, head): a_src/a_dst dot products over channel dim Cd
__global__ void k_att(const float* __restrict__ h, const float* __restrict__ att_s,
                      const float* __restrict__ att_d, float* __restrict__ a_s,
                      float* __restrict__ a_d, int N, int Cd){
  int idx = blockIdx.x * blockDim.x + threadIdx.x;
  if (idx >= N * 2) return;
  int n = idx >> 1, hd = idx & 1;
  const float* hr = h + (size_t)n * 2 * Cd + hd * Cd;
  const float* as = att_s + hd * Cd;
  const float* ad = att_d + hd * Cd;
  float s = 0.f, d = 0.f;
  for (int c = 0; c < Cd; ++c){ s += hr[c] * as[c]; d += hr[c] * ad[c]; }
  a_s[idx] = s; a_d[idx] = d;
}

__device__ __forceinline__ float edge_e(const int* src, const int* dst,
                                        const float* a_s, const float* a_d,
                                        int e, int hd){
  float v = a_s[src[e] * 2 + hd] + a_d[dst[e] * 2 + hd];
  return (v > 0.f) ? v : v * NEG_SLOPE;
}

__global__ void k_edge_max(const int* __restrict__ src, const int* __restrict__ dst,
                           const float* __restrict__ a_s, const float* __restrict__ a_d,
                           unsigned* __restrict__ m, int E){
  int idx = blockIdx.x * blockDim.x + threadIdx.x;
  if (idx >= E * 2) return;
  int e = idx >> 1, hd = idx & 1;
  float v = edge_e(src, dst, a_s, a_d, e, hd);
  atomicMax(m + dst[e] * 2 + hd, f2ord(v));
}

__global__ void k_edge_exp(const int* __restrict__ src, const int* __restrict__ dst,
                           const float* __restrict__ a_s, const float* __restrict__ a_d,
                           const unsigned* __restrict__ m, float* __restrict__ ex,
                           float* __restrict__ ssum, int E){
  int idx = blockIdx.x * blockDim.x + threadIdx.x;
  if (idx >= E * 2) return;
  int e = idx >> 1, hd = idx & 1;
  float v = edge_e(src, dst, a_s, a_d, e, hd);
  float mx = ord2f(m[dst[e] * 2 + hd]);
  float x = __expf(v - mx);
  ex[idx] = x;
  atomicAdd(ssum + dst[e] * 2 + hd, x);
}

// out[dst, f] += h[src, f] * ex[e,hd] / (ssum[dst,hd] + eps)  (thread per edge-feature)
__global__ void k_agg(const int* __restrict__ src, const int* __restrict__ dst,
                      const float* __restrict__ h, const float* __restrict__ ex,
                      const float* __restrict__ ssum, float* __restrict__ out,
                      int E, int F, int Cper){
  long long idx = (long long)blockIdx.x * blockDim.x + threadIdx.x;
  if (idx >= (long long)E * F) return;
  int e = (int)(idx / F), f = (int)(idx % F);
  int hd = f / Cper;
  int s = src[e], d = dst[e];
  float alpha = ex[e * 2 + hd] / (ssum[d * 2 + hd] + GEPS);
  atomicAdd(out + (size_t)d * F + f, h[(size_t)s * F + f] * alpha);
}

// h1 = elu(out1 + bias1) -> f16
__global__ void k_post1(const float* __restrict__ out1, const float* __restrict__ bias1,
                        _Float16* __restrict__ h1h, int total){
  int idx = blockIdx.x * blockDim.x + threadIdx.x;
  if (idx >= total) return;
  float v = out1[idx] + bias1[idx & 127];
  v = (v > 0.f) ? v : (__expf(v) - 1.f);
  h1h[idx] = (_Float16)v;
}

// hf = mean over heads of out2[N,2,128] + bias2 -> f16
__global__ void k_post2(const float* __restrict__ out2, const float* __restrict__ bias2,
                        _Float16* __restrict__ hfh, int N){
  int idx = blockIdx.x * blockDim.x + threadIdx.x;
  if (idx >= N * 128) return;
  int n = idx >> 7, c = idx & 127;
  float v = 0.5f * (out2[(size_t)n * 256 + c] + out2[(size_t)n * 256 + 128 + c]) + bias2[c];
  hfh[idx] = (_Float16)v;
}

// Edge head: 16 edges/block, 8 waves x 16 cols, K=256 gathered from hf[src]/hf[dst].
// Fused: relu(A@fc2^T + b) . fc3 + b3 -> sigmoid.
__global__ void k_edge_head(const int* __restrict__ src, const int* __restrict__ dst,
                            const _Float16* __restrict__ hf, const _Float16* __restrict__ fc2h,
                            const float* __restrict__ fc2_b, const float* __restrict__ fc3_w,
                            const float* __restrict__ fc3_b, float* __restrict__ out){
  __shared__ int s_src[16], s_dst[16];
  __shared__ float s_acc[16];
  int tid = threadIdx.x;
  int e0 = blockIdx.x * 16;
  if (tid < 16){ s_src[tid] = src[e0 + tid]; s_acc[tid] = 0.f; }
  else if (tid < 32){ s_dst[tid - 16] = dst[e0 + tid - 16]; }
  __syncthreads();
  int lane = tid & 31, wave = tid >> 5;
  int mr = lane & 15;
  int koff = (lane < 16) ? 0 : 8;
  int n = wave * 16 + mr;
  const _Float16* brow = fc2h + (size_t)n * 256;
  v8f acc = {};
  for (int k0 = 0; k0 < 256; k0 += 32){
    int klo = k0 + koff, khi = klo + 16;
    int rlo = (klo < 128) ? s_src[mr] : s_dst[mr];
    int rhi = (khi < 128) ? s_src[mr] : s_dst[mr];
    v8h alo = *(const v8h*)(hf + (size_t)rlo * 128 + (klo & 127));
    v8h ahi = *(const v8h*)(hf + (size_t)rhi * 128 + (khi & 127));
    v8h blo = *(const v8h*)(brow + klo);
    v8h bhi = *(const v8h*)(brow + khi);
    v16h a = __builtin_shufflevector(alo, ahi, 0,1,2,3,4,5,6,7,8,9,10,11,12,13,14,15);
    v16h b = __builtin_shufflevector(blo, bhi, 0,1,2,3,4,5,6,7,8,9,10,11,12,13,14,15);
    acc = __builtin_amdgcn_wmma_f32_16x16x32_f16(false, a, false, b, (short)0, acc, false, false);
  }
  float bias = fc2_b[n], w3 = fc3_w[n];
  int rowoff = (lane < 16) ? 0 : 8;
  #pragma unroll
  for (int j = 0; j < 8; ++j){
    float v = fmaxf(acc[j] + bias, 0.f);
    atomicAdd(&s_acc[rowoff + j], v * w3);   // ds_add_f32
  }
  __syncthreads();
  if (tid < 16){
    float logit = s_acc[tid] + fc3_b[0];
    out[e0 + tid] = 1.f / (1.f + __expf(-logit));
  }
}

extern "C" void kernel_launch(void* const* d_in, const int* in_sizes, int n_in,
                              void* d_out, int out_size, void* d_ws, size_t ws_size,
                              hipStream_t stream){
  const float* x      = (const float*)d_in[0];
  const int*   edge   = (const int*)d_in[1];
  const float* cls    = (const float*)d_in[2];
  const float* fc0_w  = (const float*)d_in[3];
  const float* fc0_b  = (const float*)d_in[4];
  const float* W1     = (const float*)d_in[5];
  const float* att_s1 = (const float*)d_in[6];
  const float* att_d1 = (const float*)d_in[7];
  const float* bias1  = (const float*)d_in[8];
  const float* W2     = (const float*)d_in[9];
  const float* att_s2 = (const float*)d_in[10];
  const float* att_d2 = (const float*)d_in[11];
  const float* bias2  = (const float*)d_in[12];
  const float* fc2_w  = (const float*)d_in[13];
  const float* fc2_b  = (const float*)d_in[14];
  const float* fc3_w  = (const float*)d_in[15];
  const float* fc3_b  = (const float*)d_in[16];

  const int N = in_sizes[0] / 128;
  const int E = in_sizes[1] / 2;
  const int* src = edge;
  const int* dst = edge + E;

  char* p = (char*)d_ws;
  auto alloc = [&](size_t bytes) -> void* {
    void* r = (void*)p; p += ((bytes + 255) & ~(size_t)255); return r;
  };
  float*    sent = (float*)alloc(128 * 4);
  float*    c1   = (float*)alloc(128 * 4);
  _Float16* W1t  = (_Float16*)alloc(128 * 128 * 2);
  _Float16* W2t  = (_Float16*)alloc(256 * 128 * 2);
  _Float16* fc2h = (_Float16*)alloc(128 * 256 * 2);
  _Float16* x_h  = (_Float16*)alloc((size_t)N * 128 * 2);
  float*    h    = (float*)alloc((size_t)N * 128 * 4);
  float*    a_s1 = (float*)alloc((size_t)N * 2 * 4);
  float*    a_d1 = (float*)alloc((size_t)N * 2 * 4);
  unsigned* m1   = (unsigned*)alloc((size_t)N * 2 * 4);
  float*    s1   = (float*)alloc((size_t)N * 2 * 4);
  float*    ex1  = (float*)alloc((size_t)E * 2 * 4);
  float*    out1 = (float*)alloc((size_t)N * 128 * 4);
  _Float16* h1h  = (_Float16*)alloc((size_t)N * 128 * 2);
  float*    h2   = (float*)alloc((size_t)N * 256 * 4);
  float*    a_s2 = (float*)alloc((size_t)N * 2 * 4);
  float*    a_d2 = (float*)alloc((size_t)N * 2 * 4);
  unsigned* m2   = (unsigned*)alloc((size_t)N * 2 * 4);
  float*    s2   = (float*)alloc((size_t)N * 2 * 4);
  float*    ex2  = (float*)alloc((size_t)E * 2 * 4);
  float*    out2 = (float*)alloc((size_t)N * 256 * 4);
  _Float16* hfh  = (_Float16*)alloc((size_t)N * 128 * 2);

  hipMemsetAsync(m1, 0, (size_t)N * 2 * 4, stream);
  hipMemsetAsync(s1, 0, (size_t)N * 2 * 4, stream);
  hipMemsetAsync(out1, 0, (size_t)N * 128 * 4, stream);
  hipMemsetAsync(m2, 0, (size_t)N * 2 * 4, stream);
  hipMemsetAsync(s2, 0, (size_t)N * 2 * 4, stream);
  hipMemsetAsync(out2, 0, (size_t)N * 256 * 4, stream);

  // prep: sentence proj, folded constant, weight transposes + f16 conversions
  k_sent<<<1, 128, 0, stream>>>(fc0_w, cls, fc0_b, sent);
  k_c1<<<1, 128, 0, stream>>>(W1, sent, c1);
  k_transpose_f16<<<(128 * 128 + 255) / 256, 256, 0, stream>>>(W1t, W1, 128, 128, 128 * 128);
  k_transpose_f16<<<(256 * 128 + 255) / 256, 256, 0, stream>>>(W2t, W2, 128, 256, 256 * 128);
  k_f32_to_f16<<<(128 * 256 + 255) / 256, 256, 0, stream>>>(fc2h, fc2_w, 128 * 256);
  k_f32_to_f16<<<(N * 128 + 255) / 256, 256, 0, stream>>>(x_h, x, N * 128);

  // conv1
  k_gemm<<<dim3(N / 16, 1), 256, 0, stream>>>(x_h, W1t, h, 128, 128, c1);
  k_att<<<(N * 2 + 255) / 256, 256, 0, stream>>>(h, att_s1, att_d1, a_s1, a_d1, N, 64);
  k_edge_max<<<(E * 2 + 255) / 256, 256, 0, stream>>>(src, dst, a_s1, a_d1, m1, E);
  k_edge_exp<<<(E * 2 + 255) / 256, 256, 0, stream>>>(src, dst, a_s1, a_d1, m1, ex1, s1, E);
  {
    long long tot = (long long)E * 128;
    k_agg<<<(unsigned)((tot + 255) / 256), 256, 0, stream>>>(src, dst, h, ex1, s1, out1, E, 128, 64);
  }
  k_post1<<<(N * 128 + 255) / 256, 256, 0, stream>>>(out1, bias1, h1h, N * 128);

  // conv2
  k_gemm<<<dim3(N / 16, 2), 256, 0, stream>>>(h1h, W2t, h2, 128, 256, nullptr);
  k_att<<<(N * 2 + 255) / 256, 256, 0, stream>>>(h2, att_s2, att_d2, a_s2, a_d2, N, 128);
  k_edge_max<<<(E * 2 + 255) / 256, 256, 0, stream>>>(src, dst, a_s2, a_d2, m2, E);
  k_edge_exp<<<(E * 2 + 255) / 256, 256, 0, stream>>>(src, dst, a_s2, a_d2, m2, ex2, s2, E);
  {
    long long tot = (long long)E * 256;
    k_agg<<<(unsigned)((tot + 255) / 256), 256, 0, stream>>>(src, dst, h2, ex2, s2, out2, E, 256, 128);
  }
  k_post2<<<(N * 128 + 255) / 256, 256, 0, stream>>>(out2, bias2, hfh, N);

  // edge head (WMMA, fused MLP + sigmoid)
  k_edge_head<<<E / 16, 256, 0, stream>>>(src, dst, hfh, fc2h, fc2_b, fc3_w, fc3_b, (float*)d_out);
}
